// PointNetHetero_67783173865977
// MI455X (gfx1250) — matrix operand order repaired and checked by reading, not compiled, not run on hardware
//
#include <hip/hip_runtime.h>
#include <hip/hip_bf16.h>

typedef __attribute__((ext_vector_type(16))) _Float16 v16h;
typedef __attribute__((ext_vector_type(8)))  float    v8f;

#define PPB 256
#define NB_TOTAL 2048            // 16 * 32768 / 256 blocks
#define BLOCKS_PER_CLOUD 128     // 32768 / 256
#define FLT_BIG 3.402823466e38f

// ---------------------------------------------------------------------------
// Fragment-order ("swizzled") storage: frag[tile][lane][16 halves] contiguous,
// so one fragment = one aligned 32-byte load per lane.
// B-tile (32x16):  K = kt*32 + (same lane/e -> K order as A-tile),   N = nt*16 + (lane&15)
// A-tile (16x32):  M = (lane&15);  e<8: K = (lane>>4)*8 + e
//                                  e>=8: K = 16 + (lane>>4)*8 + (e-8)
// ---------------------------------------------------------------------------
__device__ __forceinline__ v16h load_frag_swz(const _Float16* __restrict__ base,
                                              int tile, int lane) {
  return *(const v16h*)(base + (((size_t)tile * 32 + lane) << 4));
}

// A fragment from row-major LDS [M][K] with leading dim ld (ISA 7.12.2 layout)
__device__ __forceinline__ v16h load_frag_a16(const _Float16* __restrict__ tile,
                                              int ld, int lane) {
  const int m  = lane & 15;
  const int kh = (lane >> 4) << 3;          // 0 or 8
  const _Float16* r = tile + m * ld + kh;
  v16h a;
#pragma unroll
  for (int i = 0; i < 8; ++i) a[i] = r[i];          // 16B contiguous
#pragma unroll
  for (int i = 0; i < 8; ++i) a[8 + i] = r[16 + i]; // 16B contiguous
  return a;
}

// index of (block row, col) inside the swizzled A-fragment buffer for kernel 2
__device__ __forceinline__ size_t cw_swz_index(int blk, int col) {
  const int rt = blk >> 4, m = blk & 15;
  const int kt = col >> 5, cw = col & 31;
  int hi, e;
  if (cw < 16) { hi = cw >> 3; e = cw & 7; }
  else         { hi = (cw - 16) >> 3; e = 8 + ((cw - 16) & 7); }
  const int lane = (hi << 4) | m;
  return (((size_t)(rt * 24 + kt) * 32 + lane) << 4) + e;
}

// ---- prep: f32 weights -> f16 in B-fragment order -------------------------
__global__ __launch_bounds__(256) void k_prep_swz(
    const float* __restrict__ w0, const float* __restrict__ w1,
    const float* __restrict__ w2, const float* __restrict__ fw,
    _Float16* __restrict__ w0s, _Float16* __restrict__ w1s,
    _Float16* __restrict__ w2s, _Float16* __restrict__ fws) {
  const int i    = blockIdx.x * 256 + threadIdx.x;
  const int tile = i >> 9;
  const int lane = (i >> 4) & 31;
  const int e    = i & 15;
  const int krel = (e < 8) ? ((lane >> 4) << 3) + e : 16 + ((lane >> 4) << 3) + (e - 8);   // K offset within 32-wide tile
  const int nrel = lane & 15;

  if (i < 4 * 512) {                         // w0: 3x64 zero-padded to 32x64
    const int nt = tile;                     // tiles = nt (kt==0)
    w0s[i] = (krel < 3) ? (_Float16)w0[krel * 64 + nt * 16 + nrel] : (_Float16)0.f;
  }
  if (i < 16 * 512) {                        // w1: 64x128, tiles = kt*8+nt
    const int kt = tile >> 3, nt = tile & 7;
    w1s[i] = (_Float16)w1[(kt * 32 + krel) * 128 + nt * 16 + nrel];
  }
  if (i < 64 * 512) {                        // w2: 128x256, tiles = kt*16+nt
    const int kt = tile >> 4, nt = tile & 15;
    w2s[i] = (_Float16)w2[(kt * 32 + krel) * 256 + nt * 16 + nrel];
  }
  if (i < 384 * 512) {                       // fw: 768x256, tiles = kt*16+nt
    const int kt = tile >> 4, nt = tile & 15;
    fws[i] = (_Float16)fw[(kt * 32 + krel) * 256 + nt * 16 + nrel];
  }
}

// ---- kernel 1: all-WMMA pointwise MLP + fused per-block max/min/mean --------
// one workgroup (8 waves) per 256-point block; two chunks of 128 rows
__global__ __launch_bounds__(256) void k_point_mlp(
    const float* __restrict__ data,
    const _Float16* __restrict__ w0s, const float* __restrict__ b0,
    const _Float16* __restrict__ w1s, const float* __restrict__ b1,
    const _Float16* __restrict__ w2s, const float* __restrict__ b2,
    _Float16* __restrict__ cws) {
  __shared__ _Float16 sh0[128 * 64];      // 16 KB  layer-0 activations (f16)
  __shared__ _Float16 shbuf[128 * 128];   // 32 KB  pnts staging + layer-1 acts
  __shared__ float    sstat[3 * 8 * 256]; // 24 KB  wave-private stat slices

  _Float16* spnt = shbuf;                 // [128][32] zero-padded pnts tile
  _Float16* sh1  = shbuf;                 // [128][128] layer-1 activations
  float* smax = sstat;
  float* smin = sstat + 2048;
  float* ssum = sstat + 4096;

  const int tid  = threadIdx.x;
  const int lane = tid & 31;
  const int wv   = tid >> 5;              // 8 waves
  const int blk  = blockIdx.x;
  const int m0   = wv * 16;               // chunk-local M-tile for this wave
  const int nlo  = lane & 15;
  const int rhi  = (lane >> 4) << 3;      // 0 or 8: C-row offset of this lane

  // init wave-private stat accumulators
#pragma unroll
  for (int w = 0; w < 8; ++w) {
    smax[w * 256 + tid] = -FLT_BIG;
    smin[w * 256 + tid] =  FLT_BIG;
    ssum[w * 256 + tid] =  0.f;
  }
  __syncthreads();

  for (int chunk = 0; chunk < 2; ++chunk) {
    const int rbase = chunk * 128;

    // ---- stage pnts = (xyz - center) * scale into zero-padded A tile -------
    {
      const int row  = tid >> 1;            // two threads per row
      const int half = tid & 1;             // low/high 16 K-cols
      v16h z = {};
      if (half == 0) {
        const float* x = data + ((size_t)blk * PPB + rbase + row) * 9;
        z[0] = (_Float16)((x[0] - x[3]) * x[6]);
        z[1] = (_Float16)((x[1] - x[4]) * x[6]);
        z[2] = (_Float16)((x[2] - x[5]) * x[6]);
      }
      *(v16h*)(spnt + row * 32 + half * 16) = z;
    }
    __syncthreads();

    // ---- layer 0 WMMA: [128x32(pad)]@[32x64] + b0, ReLU -> sh0 -------------
    {
      v16h a0 = load_frag_a16(spnt + m0 * 32, 32, lane);
#pragma unroll
      for (int nt = 0; nt < 4; ++nt) {
        v16h b = load_frag_swz(w0s, nt, lane);
        v8f acc = {};
        acc = __builtin_amdgcn_wmma_f32_16x16x32_f16(false, a0, false, b,
                                                     (short)0, acc, false, false);
        const int ncol = nt * 16 + nlo;
        const float bias = b0[ncol];
#pragma unroll
        for (int r = 0; r < 8; ++r)
          sh0[(m0 + rhi + r) * 64 + ncol] = (_Float16)fmaxf(acc[r] + bias, 0.f);
      }
    }
    __syncthreads();   // spnt (aliases sh1) now dead for all waves

    // ---- layer 1 WMMA: [128x64]@[64x128] + b1, ReLU -> sh1 -----------------
    {
      v16h a1[2];
#pragma unroll
      for (int kt = 0; kt < 2; ++kt)
        a1[kt] = load_frag_a16(sh0 + m0 * 64 + kt * 32, 64, lane);
#pragma unroll 2
      for (int nt = 0; nt < 8; ++nt) {
        v8f acc = {};
#pragma unroll
        for (int kt = 0; kt < 2; ++kt) {
          v16h b = load_frag_swz(w1s, kt * 8 + nt, lane);
          acc = __builtin_amdgcn_wmma_f32_16x16x32_f16(false, a1[kt], false, b,
                                                       (short)0, acc, false, false);
        }
        const int ncol = nt * 16 + nlo;
        const float bias = b1[ncol];
#pragma unroll
        for (int r = 0; r < 8; ++r)
          sh1[(m0 + rhi + r) * 128 + ncol] = (_Float16)fmaxf(acc[r] + bias, 0.f);
      }
    }
    // sh1 rows are wave-local: no barrier needed before layer 2

    // ---- layer 2 WMMA: [128x128]@[128x256], fused segment max/min/sum ------
    {
      v16h a2[4];
#pragma unroll
      for (int kt = 0; kt < 4; ++kt)
        a2[kt] = load_frag_a16(sh1 + m0 * 128 + kt * 32, 128, lane);
#pragma unroll 2
      for (int nt = 0; nt < 16; ++nt) {
        v8f acc = {};
#pragma unroll
        for (int kt = 0; kt < 4; ++kt) {
          v16h b = load_frag_swz(w2s, kt * 16 + nt, lane);
          acc = __builtin_amdgcn_wmma_f32_16x16x32_f16(false, a2[kt], false, b,
                                                       (short)0, acc, false, false);
        }
        // per-lane reduce over the 8 C rows this lane holds
        float lmax = -FLT_BIG, lmin = FLT_BIG, lsum = 0.f;
#pragma unroll
        for (int r = 0; r < 8; ++r) {
          lmax = fmaxf(lmax, acc[r]);
          lmin = fminf(lmin, acc[r]);
          lsum += acc[r];
        }
        // combine lane n with lane n+16 (other 8 rows of the tile)
        lmax = fmaxf(lmax, __shfl_xor(lmax, 16, 32));
        lmin = fminf(lmin, __shfl_xor(lmin, 16, 32));
        lsum += __shfl_xor(lsum, 16, 32);
        if (lane < 16) {                    // wave-private LDS accumulate
          const int ix = wv * 256 + nt * 16 + lane;
          smax[ix] = fmaxf(smax[ix], lmax);
          smin[ix] = fminf(smin[ix], lmin);
          ssum[ix] += lsum;
        }
      }
    }
    __syncthreads();   // before spnt/sh0 reuse in next chunk
  }

  // ---- deterministic cross-wave reduction (stats already in LDS) ----------
  float vmx = smax[tid], vmn = smin[tid], vsm = ssum[tid];
#pragma unroll
  for (int w = 1; w < 8; ++w) {
    vmx = fmaxf(vmx, smax[w * 256 + tid]);
    vmn = fminf(vmn, smin[w * 256 + tid]);
    vsm += ssum[w * 256 + tid];
  }
  const float bc = b2[tid];
  const float omax  = vmx + bc;
  const float omin  = vmn + bc;
  const float omean = vsm * (1.0f / (float)PPB) + bc;

  // scatter straight into kernel-2's A-fragment layout:
  // cw row = blk, cols [max: tid | min: 256+tid | mean: 512+tid]
  cws[cw_swz_index(blk, tid)]       = (_Float16)omax;
  cws[cw_swz_index(blk, 256 + tid)] = (_Float16)omin;
  cws[cw_swz_index(blk, 512 + tid)] = (_Float16)omean;
}

// ---- kernel 2: block_feature = cw_inp @ fw + fb, plus metadata columns ------
// one workgroup per 16 block-rows; wave wv owns output cols [wv*32, wv*32+32)
__global__ __launch_bounds__(256) void k_block_fc(
    const _Float16* __restrict__ cws, const _Float16* __restrict__ fws,
    const float* __restrict__ fb, const float* __restrict__ data,
    float* __restrict__ out) {
  const int tid  = threadIdx.x;
  const int lane = tid & 31;
  const int wv   = tid >> 5;
  const int rt   = blockIdx.x;        // block-row tile (16 rows)
  const int r0   = rt * 16;
  const int nt0  = wv * 2;            // two N-tiles per wave

  v8f acc0 = {}, acc1 = {};
#pragma unroll 2
  for (int kt = 0; kt < 24; ++kt) {   // K = 768 = 24 x 32
    const int ktn = (kt < 23) ? kt + 1 : 23;
    __builtin_prefetch(fws + ((size_t)(ktn * 16 + nt0) << 9), 0, 1);
    v16h a   = load_frag_swz(cws, rt * 24 + kt, lane);
    v16h bf0 = load_frag_swz(fws, kt * 16 + nt0, lane);
    v16h bf1 = load_frag_swz(fws, kt * 16 + nt0 + 1, lane);
    acc0 = __builtin_amdgcn_wmma_f32_16x16x32_f16(false, a, false, bf0,
                                                  (short)0, acc0, false, false);
    acc1 = __builtin_amdgcn_wmma_f32_16x16x32_f16(false, a, false, bf1,
                                                  (short)0, acc1, false, false);
  }

  const int nlo  = lane & 15;
  const int rrow = r0 + ((lane >> 4) << 3);
  {
    const int c = nt0 * 16 + nlo;
    const float bias = fb[c];
#pragma unroll
    for (int r = 0; r < 8; ++r)
      out[(size_t)(rrow + r) * 262 + c] = acc0[r] + bias;
  }
  {
    const int c = (nt0 + 1) * 16 + nlo;
    const float bias = fb[c];
#pragma unroll
    for (int r = 0; r < 8; ++r)
      out[(size_t)(rrow + r) * 262 + c] = acc1[r] + bias;
  }

  // metadata: [cw(256) | pntcnt | center(3) | scale | pc_start]
  if (tid < 16) {
    const int b = r0 + tid;
    const float* x = data + (size_t)b * PPB * 9;  // first row of the block
    float* o = out + (size_t)b * 262;
    o[256] = x[7];                  // pntcnt
    o[257] = x[3];                  // center.x
    o[258] = x[4];                  // center.y
    o[259] = x[5];                  // center.z
    o[260] = x[6];                  // scale
    o[261] = (b % BLOCKS_PER_CLOUD == 0) ? 1.0f : 0.0f;  // pc_start
  }
}

extern "C" void kernel_launch(void* const* d_in, const int* in_sizes, int n_in,
                              void* d_out, int out_size, void* d_ws, size_t ws_size,
                              hipStream_t stream) {
  (void)in_sizes; (void)n_in; (void)out_size; (void)ws_size;
  const float* data = (const float*)d_in[0];
  const float* w0   = (const float*)d_in[1];
  const float* b0   = (const float*)d_in[2];
  const float* w1   = (const float*)d_in[3];
  const float* b1   = (const float*)d_in[4];
  const float* w2   = (const float*)d_in[5];
  const float* b2   = (const float*)d_in[6];
  const float* fw   = (const float*)d_in[7];
  const float* fb   = (const float*)d_in[8];

  // workspace (f16, all fragment-order):
  //   cws[128 row-tiles * 24 k-tiles * 512] | w0s[4*512] | w1s[16*512]
  //   | w2s[64*512] | fws[384*512]
  _Float16* cws = (_Float16*)d_ws;
  _Float16* w0s = cws + (size_t)(NB_TOTAL / 16) * 24 * 512;
  _Float16* w1s = w0s + 4 * 512;
  _Float16* w2s = w1s + 16 * 512;
  _Float16* fws = w2s + 64 * 512;

  k_prep_swz<<<(384 * 512 + 255) / 256, 256, 0, stream>>>(w0, w1, w2, fw,
                                                          w0s, w1s, w2s, fws);
  k_point_mlp<<<NB_TOTAL, 256, 0, stream>>>(data, w0s, b0, w1s, b1, w2s, b2, cws);
  k_block_fc<<<NB_TOTAL / 16, 256, 0, stream>>>(cws, fws, fb, data, (float*)d_out);
}
